// Mamba2HKHV_8486855377331
// MI455X (gfx1250) — compile-verified, matrix operand order checked by reading, not run on hardware
//
#include <hip/hip_runtime.h>
#include <hip/hip_bf16.h>

// ---------------------------------------------------------------------------
// Mamba2 block + head projection for MI455X (gfx1250, wave32, WMMA).
//   1) cvt f32->f16 (X, in_proj_w, out_proj_w, to_h_w)
//   2) WMMA GEMM: zxbcdt = X @ in_proj_w^T            (8192 x 4256, K=1024)
//   3) dt = softplus(dt_raw + dt_bias)
//   4) causal depthwise conv (taps=4) + bias + SiLU -> xBC_act
//   5) recurrent SSM scan, 64x64 state per (b,head) in registers,
//      async-LDS double-buffered operand staging
//   6) y * silu(z), RMSNorm, -> f16
//   7) WMMA GEMM: H = yn @ out_proj_w^T (f16 out)     (8192 x 1024, K=2048)
//   8) WMMA GEMM: HKV = H @ to_h_w^T, split into HK | HV in d_out
// GEMM: 256x128 block tile, 64x64 per wave (4x4 WMMA fragments, 16 v8f
// accumulators), GLOBAL_LOAD_ASYNC_TO_LDS_B128 double buffering (ASYNCcnt),
// one barrier per K-iteration.
// ---------------------------------------------------------------------------

typedef __attribute__((ext_vector_type(16))) _Float16 v16h;
typedef __attribute__((ext_vector_type(8)))  _Float16 v8h;
typedef __attribute__((ext_vector_type(8)))  float    v8f;

#define BATCH     2
#define SEQLEN    4096
#define DIMM      1024
#define D_INNER   2048
#define NHEADS    32
#define HEADDIM   64
#define D_STATE   64
#define CONV_DIM  2176           // D_INNER + 2*D_STATE
#define D_IN_PROJ 4256           // 2*D_INNER + 2*D_STATE + NHEADS
#define MTOT      (BATCH*SEQLEN) // 8192 rows
#define EPS_RMS   1e-5f

#if __has_builtin(__builtin_amdgcn_global_load_async_to_lds_b128) && \
    __has_builtin(__builtin_amdgcn_global_load_async_to_lds_b32)
#define HAVE_ASYNC_LDS 1
#else
#define HAVE_ASYNC_LDS 0
#endif

#if HAVE_ASYNC_LDS
// exact parameter types per clang diagnostic:
//   b128: (int __vector(4) addrspace(1)*, int __vector(4) addrspace(3)*, Imm, Imm)
//   b32 : (int addrspace(1)*,             int addrspace(3)*,             Imm, Imm)
typedef int v4i_g __attribute__((__vector_size__(16)));
typedef __attribute__((address_space(1))) v4i_g* gv4i_p;
typedef __attribute__((address_space(3))) v4i_g* lv4i_p;
typedef __attribute__((address_space(1))) int*   gi32_p;
typedef __attribute__((address_space(3))) int*   li32_p;
#endif

// global -> LDS copy, async (ASYNCcnt) via gfx1250 builtins
__device__ __forceinline__ void cp16_g2l(void* lds, const void* g) {
#if HAVE_ASYNC_LDS
  __builtin_amdgcn_global_load_async_to_lds_b128((gv4i_p)g, (lv4i_p)lds, 0, 0);
#else
  *(v8h*)lds = *(const v8h*)g;
#endif
}
__device__ __forceinline__ void cp4_g2l(void* lds, const void* g) {
#if HAVE_ASYNC_LDS
  __builtin_amdgcn_global_load_async_to_lds_b32((gi32_p)g, (li32_p)lds, 0, 0);
#else
  *(float*)lds = *(const float*)g;
#endif
}
__device__ __forceinline__ void wait_async_lds() {
#if HAVE_ASYNC_LDS
#if __has_builtin(__builtin_amdgcn_s_wait_asynccnt)
  __builtin_amdgcn_s_wait_asynccnt(0);
#else
  asm volatile("s_wait_asynccnt 0x0" ::: "memory");
#endif
#endif
}

// ---------------------------------------------------------------- converts
__global__ void k_cvt_f16(const float* __restrict__ in, _Float16* __restrict__ out, int n) {
  int i = blockIdx.x * blockDim.x + threadIdx.x;
  if (i < n) out[i] = (_Float16)in[i];
}

// ---------------------------------------------------------------- WMMA GEMM
// C[M,N] = A[M,K] * B[N,K]^T ; A,B f16 row-major (K contiguous).
// ISA 16-bit operand layout: lane m%16; lanes 0-15 hold K {0..7,16..23},
// lanes 16-31 hold K {8..15,24..31}  -> two b128 LDS reads per fragment.
__device__ __forceinline__ v16h load_frag(const _Float16* s, int row0, int kbase, int lane) {
  const int r   = row0 + (lane & 15);
  const int sel = (lane >> 4) << 3;
  v8h lo = *(const v8h*)(s + r * 64 + kbase + sel);
  v8h hi = *(const v8h*)(s + r * 64 + kbase + 16 + sel);
  v16h f;
#pragma unroll
  for (int j = 0; j < 8; ++j) { f[j] = lo[j]; f[j + 8] = hi[j]; }
  return f;
}

__device__ __forceinline__ void store_tile(const v8f& c, int m0, int n0, int N,
                                           float* C0, int ldc0,
                                           float* C1, int ldc1, int nsplit,
                                           _Float16* Ch, int ldch) {
  if (n0 >= N) return;
#pragma unroll
  for (int r = 0; r < 8; ++r) {
    const int m = m0 + r;
    const float v = c[r];
    if (C0) {
      if (n0 < nsplit) C0[(size_t)m * ldc0 + n0] = v;
      else             C1[(size_t)m * ldc1 + (n0 - nsplit)] = v;
    }
    if (Ch) Ch[(size_t)m * ldch + n0] = (_Float16)v;
  }
}

__launch_bounds__(256)
__global__ void k_gemm_f16(const _Float16* __restrict__ A, const _Float16* __restrict__ Bw,
                           int N, int K, int lda, int ldb,
                           float* __restrict__ C0, int ldc0,
                           float* __restrict__ C1, int ldc1, int nsplit,
                           _Float16* __restrict__ Ch, int ldch) {
  __shared__ alignas(16) _Float16 As[2][256 * 64];  // 2 x 32 KB
  __shared__ alignas(16) _Float16 Bs[2][128 * 64];  // 2 x 16 KB

  const int tid  = threadIdx.x;
  const int lane = tid & 31;
  const int wid  = tid >> 5;            // 8 waves (wave32)
  const int wm   = (wid & 3) * 64;      // 4 wave-rows
  const int wn   = (wid >> 2) * 64;     // 2 wave-cols
  const int mblock = blockIdx.y * 256;
  const int nblock = blockIdx.x * 128;

  // stage one 256x64 A tile + 128x64 B tile into LDS buffer `buf`
  auto stage = [&](int buf, int k0) {
#pragma unroll
    for (int i = 0; i < 8; ++i) {               // A: 8 x b128 per thread
      const int lin = tid + i * 256;
      const int row = lin >> 3;
      const int col = (lin & 7) << 3;
      cp16_g2l(&As[buf][row * 64 + col],
               A + (size_t)(mblock + row) * lda + k0 + col);
    }
#pragma unroll
    for (int i = 0; i < 4; ++i) {               // B: 4 x b128 per thread
      const int lin = tid + i * 256;
      const int row = lin >> 3;
      const int col = (lin & 7) << 3;
      const int n   = nblock + row;
      if (n < N) cp16_g2l(&Bs[buf][row * 64 + col],
                          Bw + (size_t)n * ldb + k0 + col);
      else       *(v8h*)&Bs[buf][row * 64 + col] = (v8h){};  // zero-pad
    }
  };

  v8f acc[4][4];
#pragma unroll
  for (int i = 0; i < 4; ++i)
#pragma unroll
    for (int j = 0; j < 4; ++j) acc[i][j] = (v8f){};

  stage(0, 0);
  wait_async_lds();
  __syncthreads();

  int cur = 0;
  for (int k0 = 0; k0 < K; k0 += 64) {
    if (k0 + 64 < K) stage(cur ^ 1, k0 + 64);   // async prefetch next tile
#pragma unroll
    for (int ks = 0; ks < 2; ++ks) {
      v16h afr[4], bfr[4];
#pragma unroll
      for (int i = 0; i < 4; ++i) afr[i] = load_frag(As[cur], wm + i * 16, ks * 32, lane);
#pragma unroll
      for (int j = 0; j < 4; ++j) bfr[j] = load_frag(Bs[cur], wn + j * 16, ks * 32, lane);
#pragma unroll
      for (int i = 0; i < 4; ++i)
#pragma unroll
        for (int j = 0; j < 4; ++j)
          acc[i][j] = __builtin_amdgcn_wmma_f32_16x16x32_f16(
              false, afr[i], false, bfr[j], (short)0, acc[i][j], false, false);
    }
    wait_async_lds();      // my async fills for cur^1 are done
    __syncthreads();       // everyone done reading cur / filling cur^1
    cur ^= 1;
  }

  // C/D layout: VGPR r -> M = r + 8*(lane/16), N = lane%16
  const int mrow = mblock + wm + ((lane >> 4) << 3);
  const int ncol = nblock + wn + (lane & 15);
#pragma unroll
  for (int i = 0; i < 4; ++i)
#pragma unroll
    for (int j = 0; j < 4; ++j)
      store_tile(acc[i][j], mrow + i * 16, ncol + j * 16, N,
                 C0, ldc0, C1, ldc1, nsplit, Ch, ldch);
}

// ---------------------------------------------------------------- dt = softplus
__global__ void k_dt(const float* __restrict__ zx, const float* __restrict__ dt_bias,
                     float* __restrict__ dt) {
  const int i = blockIdx.x * blockDim.x + threadIdx.x; // m*32 + h
  if (i >= MTOT * NHEADS) return;
  const int m = i >> 5, h = i & 31;
  const float x = zx[(size_t)m * D_IN_PROJ + D_INNER + CONV_DIM + h] + dt_bias[h];
  dt[i] = (x > 20.f) ? x : log1pf(__expf(x));
}

// ---------------------------------------------------------------- conv + SiLU
__global__ void k_conv(const float* __restrict__ zx, const float* __restrict__ cw,
                       const float* __restrict__ cb, float* __restrict__ out) {
  const long i = (long)blockIdx.x * blockDim.x + threadIdx.x;
  if (i >= (long)MTOT * CONV_DIM) return;
  const int m = (int)(i / CONV_DIM), c = (int)(i % CONV_DIM);
  const int b = m >> 12, l = m & 4095;
  float acc = cb[c];
#pragma unroll
  for (int j = 0; j < 4; ++j) {
    const int ll = l - 3 + j;
    if (ll >= 0)
      acc += cw[c * 4 + j] * zx[(size_t)(b * SEQLEN + ll) * D_IN_PROJ + D_INNER + c];
  }
  out[i] = acc / (1.f + __expf(-acc)); // SiLU
}

// ---------------------------------------------------------------- SSM scan
// One block per (batch, head). 256 threads: thread = (p = tid>>2, nc = tid&3).
// State h[64][64] in registers (16 floats/thread). Operands for step l+1 are
// async-prefetched into the alternate LDS buffer while step l computes.
__launch_bounds__(256)
__global__ void k_scan(const float* __restrict__ dt, const float* __restrict__ xbc,
                       const float* __restrict__ A_log, const float* __restrict__ Dp,
                       float* __restrict__ y) {
  const int b  = blockIdx.x >> 5;
  const int hh = blockIdx.x & 31;
  const int tid = threadIdx.x;
  const int nc = tid & 3, p = tid >> 2, n0 = nc << 4;
  __shared__ float sx[2][64], sB[2][64], sC[2][64], sdt[2];

  const float A  = -__expf(A_log[hh]);
  const float Dh = Dp[hh];
  float h[16];
#pragma unroll
  for (int j = 0; j < 16; ++j) h[j] = 0.f;

  auto stage = [&](int buf, int l) {
    const size_t row = (size_t)(b * SEQLEN + l);
    if (tid < 64)        cp4_g2l(&sx[buf][tid],        &xbc[row * CONV_DIM + hh * 64 + tid]);
    else if (tid < 128)  cp4_g2l(&sB[buf][tid - 64],   &xbc[row * CONV_DIM + D_INNER + (tid - 64)]);
    else if (tid < 192)  cp4_g2l(&sC[buf][tid - 128],  &xbc[row * CONV_DIM + D_INNER + D_STATE + (tid - 128)]);
    else if (tid == 192) cp4_g2l(&sdt[buf],            &dt[row * NHEADS + hh]);
  };

  stage(0, 0);
  wait_async_lds();
  __syncthreads();

  int buf = 0;
  for (int l = 0; l < SEQLEN; ++l) {
    if (l + 1 < SEQLEN) stage(buf ^ 1, l + 1);   // async prefetch next step

    const float dtv  = sdt[buf];
    const float dA   = __expf(dtv * A);
    const float xp   = sx[buf][p];
    const float coef = dtv * xp;
    float acc = 0.f;
#pragma unroll
    for (int j = 0; j < 16; ++j) {
      h[j] = h[j] * dA + coef * sB[buf][n0 + j];
      acc += h[j] * sC[buf][n0 + j];
    }
    // reduce over the 4 n-chunks (lanes nc=0..3 adjacent within the wave)
    acc += __shfl_xor(acc, 1, 32);
    acc += __shfl_xor(acc, 2, 32);
    if (nc == 0)
      y[(size_t)(b * SEQLEN + l) * D_INNER + hh * 64 + p] = acc + xp * Dh;

    wait_async_lds();
    __syncthreads();
    buf ^= 1;
  }
}

// ---------------------------------------------------------------- gate + RMSNorm
__launch_bounds__(256)
__global__ void k_gate_norm(const float* __restrict__ y, const float* __restrict__ zx,
                            const float* __restrict__ nw, _Float16* __restrict__ yn) {
  const int m = blockIdx.x;
  const int tid = threadIdx.x;
  __shared__ float red[8];
  float g[8];
  float ss = 0.f;
#pragma unroll
  for (int i = 0; i < 8; ++i) {
    const int idx = i * 256 + tid;
    const float z = zx[(size_t)m * D_IN_PROJ + idx];
    const float v = y[(size_t)m * D_INNER + idx] * (z / (1.f + __expf(-z)));
    g[i] = v;
    ss += v * v;
  }
#pragma unroll
  for (int o = 16; o > 0; o >>= 1) ss += __shfl_xor(ss, o, 32);
  if ((tid & 31) == 0) red[tid >> 5] = ss;
  __syncthreads();
  float tot = 0.f;
#pragma unroll
  for (int w = 0; w < 8; ++w) tot += red[w];
  const float scale = rsqrtf(tot / (float)D_INNER + EPS_RMS);
#pragma unroll
  for (int i = 0; i < 8; ++i) {
    const int idx = i * 256 + tid;
    yn[(size_t)m * D_INNER + idx] = (_Float16)(g[i] * scale * nw[idx]);
  }
}

// ---------------------------------------------------------------- launcher
extern "C" void kernel_launch(void* const* d_in, const int* in_sizes, int n_in,
                              void* d_out, int out_size, void* d_ws, size_t ws_size,
                              hipStream_t stream) {
  const float* X        = (const float*)d_in[0];
  const float* in_projw = (const float*)d_in[1];
  const float* conv_w   = (const float*)d_in[2];
  const float* conv_b   = (const float*)d_in[3];
  const float* dt_bias  = (const float*)d_in[4];
  const float* A_log    = (const float*)d_in[5];
  const float* Dp       = (const float*)d_in[6];
  const float* norm_w   = (const float*)d_in[7];
  const float* out_projw= (const float*)d_in[8];
  const float* to_h_w   = (const float*)d_in[9];

  float* HK = (float*)d_out;
  float* HV = HK + (size_t)MTOT * DIMM;

  char* ws = (char*)d_ws;
  size_t off = 0;
  auto carve = [&](size_t bytes) -> char* {
    char* p = ws + off;
    off = (off + bytes + 255) & ~(size_t)255;
    return p;
  };
  _Float16* Xh     = (_Float16*)carve((size_t)MTOT * DIMM * 2);
  _Float16* W1h    = (_Float16*)carve((size_t)D_IN_PROJ * DIMM * 2);
  _Float16* W2h    = (_Float16*)carve((size_t)DIMM * D_INNER * 2);
  _Float16* W3h    = (_Float16*)carve((size_t)(2 * DIMM) * DIMM * 2);
  float*    zx     = (float*)   carve((size_t)MTOT * D_IN_PROJ * 4);
  float*    xbc    = (float*)   carve((size_t)MTOT * CONV_DIM * 4);
  float*    dtbuf  = (float*)   carve((size_t)MTOT * NHEADS * 4);
  float*    ybuf   = (float*)   carve((size_t)MTOT * D_INNER * 4);
  _Float16* ynh    = (_Float16*)carve((size_t)MTOT * D_INNER * 2);
  _Float16* Hh     = (_Float16*)carve((size_t)MTOT * DIMM * 2);

  {
    int n;
    n = MTOT * DIMM;      k_cvt_f16<<<(n + 255) / 256, 256, 0, stream>>>(X,         Xh,  n);
    n = D_IN_PROJ * DIMM; k_cvt_f16<<<(n + 255) / 256, 256, 0, stream>>>(in_projw,  W1h, n);
    n = DIMM * D_INNER;   k_cvt_f16<<<(n + 255) / 256, 256, 0, stream>>>(out_projw, W2h, n);
    n = 2 * DIMM * DIMM;  k_cvt_f16<<<(n + 255) / 256, 256, 0, stream>>>(to_h_w,    W3h, n);
  }

  { // zxbcdt = X @ in_proj_w^T
    dim3 grid((D_IN_PROJ + 127) / 128, MTOT / 256);
    k_gemm_f16<<<grid, 256, 0, stream>>>(Xh, W1h, D_IN_PROJ, DIMM, DIMM, DIMM,
                                         zx, D_IN_PROJ, nullptr, 0, D_IN_PROJ,
                                         nullptr, 0);
  }

  k_dt<<<(MTOT * NHEADS + 255) / 256, 256, 0, stream>>>(zx, dt_bias, dtbuf);

  {
    long n = (long)MTOT * CONV_DIM;
    k_conv<<<(unsigned)((n + 255) / 256), 256, 0, stream>>>(zx, conv_w, conv_b, xbc);
  }

  k_scan<<<BATCH * NHEADS, 256, 0, stream>>>(dtbuf, xbc, A_log, Dp, ybuf);

  k_gate_norm<<<MTOT, 256, 0, stream>>>(ybuf, zx, norm_w, ynh);

  { // H = yn @ out_proj_w^T (f16 out)
    dim3 grid((DIMM + 127) / 128, MTOT / 256);
    k_gemm_f16<<<grid, 256, 0, stream>>>(ynh, W2h, DIMM, D_INNER, D_INNER, D_INNER,
                                         nullptr, 0, nullptr, 0, DIMM,
                                         Hh, DIMM);
  }

  { // HKV = H @ to_h_w^T, split HK | HV into d_out
    dim3 grid((2 * DIMM + 127) / 128, MTOT / 256);
    k_gemm_f16<<<grid, 256, 0, stream>>>(Hh, W3h, 2 * DIMM, DIMM, DIMM, DIMM,
                                         HK, DIMM, HV, DIMM, DIMM,
                                         nullptr, 0);
  }
}
